// CausalSelfAttention_77773267796675
// MI455X (gfx1250) — compile-verified, matrix operand order checked by reading, not run on hardware
//
#include <hip/hip_runtime.h>
#include <hip/hip_bf16.h>

#define Bn 2
#define Tn 2048
#define Cn 1024
#define Hn 16
#define HDn 64

typedef __bf16 bf16_t;
typedef unsigned int u32;
typedef __attribute__((ext_vector_type(16))) __bf16 v16bf;
typedef __attribute__((ext_vector_type(8)))  __bf16 v8bf;
typedef __attribute__((ext_vector_type(8)))  float  v8f;
typedef __attribute__((ext_vector_type(4)))  float  v4f;
typedef __attribute__((ext_vector_type(4)))  u32    v4u;
typedef __attribute__((ext_vector_type(8)))  int    v8i;
typedef __attribute__((ext_vector_type(4)))  int    v4i;

__device__ __forceinline__ v16bf combine16(v8bf lo, v8bf hi) {
    union { v16bf v; v8bf h[2]; } u;
    u.h[0] = lo; u.h[1] = hi;
    return u.v;
}

// ---------------------------------------------------------------------------
// TDM: load a 64-row x 128-byte bf16 tile (row stride = rstride_dw DWORDs)
// into LDS at byte offset lds_off, padding each 32-DWORD row to 40 DWORDs
// (-> LDS row stride of 80 bf16 elements).  D# per CDNA5 ISA.
// 6-arg builtin form (amdgpu-toolchain / clang-23 signature).
// ---------------------------------------------------------------------------
__device__ __forceinline__ void tdm_load_k_tile(
    u32 lds_off, const bf16_t* gptr, u32 rstride_dw)
{
    unsigned long long ga = (unsigned long long)(uintptr_t)gptr;
    v4u g0;
    g0[0] = 1u;                                   // count=1, user descriptor
    g0[1] = lds_off;                              // lds_addr (bytes)
    g0[2] = (u32)ga;                              // global_addr[31:0]
    g0[3] = (u32)((ga >> 32) & 0x01FFFFFFu)       // global_addr[56:32]
          | (2u << 30);                           // type = 2 ("image")
    v8i g1;
    g1[0] = (int)((2u << 16)                      // data_size = 4 bytes
                | (1u << 20)                      // pad_enable
                | (4u << 22)                      // pad_interval: 32 DWORDs
                | (7u << 25));                    // pad_amount:   8 DWORDs
    g1[1] = (int)(32u << 16);                     // tensor_dim0 = 32 DW (lo16)
    g1[2] = (int)(64u << 16);                     // dim0 hi=0 | tensor_dim1=64 (lo16)
    g1[3] = (int)(32u << 16);                     // dim1 hi=0 | tile_dim0 = 32 DW
    g1[4] = (int)(64u);                           // tile_dim1 = 64 rows, tile_dim2=0
    g1[5] = (int)rstride_dw;                      // tensor_dim0_stride[31:0]
    g1[6] = 0;                                    // stride hi | dim1_stride lo
    g1[7] = 0;
    v4i z4 = {0, 0, 0, 0};
    v8i z8 = {0, 0, 0, 0, 0, 0, 0, 0};
    __builtin_amdgcn_tensor_load_to_lds(g0, g1, z4, z4, z8, 0);
}

// ---------------------------------------------------------------------------
// Generic tiled GEMM: C[M,N] = A[M,K] * B[K,N] (+bias), bf16 WMMA, f32 accum.
// Block: 256 threads = 8 waves, tile 128x128, K-step 32.
// Waves: 4 along M x 2 along N; each wave -> 32x64 = 2x4 WMMA tiles.
// ---------------------------------------------------------------------------
template<int A_BF16, int HAS_BIAS, int OUT_BF16>
__global__ __launch_bounds__(256) void gemm_wmma_k(
    const void* __restrict__ Aptr, const float* __restrict__ Bptr,
    const float* __restrict__ bias, void* __restrict__ Cptr,
    int M, int N, int K)
{
    __shared__ bf16_t Ash[128][40];   // [m][k], row stride 80B (16B multiple)
    __shared__ bf16_t Bsh[128][40];   // transposed: [n][k]

    const int tid   = threadIdx.x;
    const int wave  = tid >> 5;
    const int lane  = tid & 31;
    const int ln    = lane & 15;
    const int lg    = lane >> 4;
    const int waveM = wave & 3;       // 0..3
    const int waveN = wave >> 2;      // 0..1
    const int m0 = blockIdx.y * 128;
    const int n0 = blockIdx.x * 128;

    v8f acc[2][4] = {};

    for (int k0 = 0; k0 < K; k0 += 32) {
        __syncthreads();
        // ---- stage A tile (128x32) into LDS as bf16 ----
        if (A_BF16) {
            const bf16_t* A = (const bf16_t*)Aptr;
            #pragma unroll
            for (int p = 0; p < 2; ++p) {
                int row = (tid >> 2) + p * 64;
                int kk  = (tid & 3) * 8;
                v8bf av = *(const v8bf*)(A + (size_t)(m0 + row) * K + k0 + kk);
                *(v8bf*)&Ash[row][kk] = av;
            }
            if (k0 + 32 < K)
                __builtin_prefetch((const bf16_t*)Aptr +
                    (size_t)(m0 + (tid >> 2)) * K + k0 + 32 + (tid & 3) * 8, 0, 0);
        } else {
            const float* A = (const float*)Aptr;
            #pragma unroll
            for (int p = 0; p < 4; ++p) {
                int row = (tid >> 3) + p * 32;
                int kk  = (tid & 7) * 4;
                v4f av = *(const v4f*)(A + (size_t)(m0 + row) * K + k0 + kk);
                #pragma unroll
                for (int i = 0; i < 4; ++i) Ash[row][kk + i] = (bf16_t)av[i];
            }
            if (k0 + 32 < K)
                __builtin_prefetch((const float*)Aptr +
                    (size_t)(m0 + (tid >> 3)) * K + k0 + 32 + (tid & 7) * 4, 0, 0);
        }
        // ---- stage B tile (32x128) into LDS transposed [n][k] ----
        #pragma unroll
        for (int p = 0; p < 4; ++p) {
            int kk = (tid >> 5) + p * 8;      // 0..31
            int nn = (tid & 31) * 4;          // 0..124
            v4f bv = *(const v4f*)(Bptr + (size_t)(k0 + kk) * N + n0 + nn);
            #pragma unroll
            for (int i = 0; i < 4; ++i) Bsh[nn + i][kk] = (bf16_t)bv[i];
        }
        if (k0 + 32 < K)
            __builtin_prefetch(Bptr + (size_t)(k0 + 32 + (tid >> 5)) * N +
                               n0 + (tid & 31) * 4, 0, 0);
        __syncthreads();

        // ---- fragments: lane m = ln, half g = lg; chunks k=g*8 and 16+g*8 ----
        v16bf afrag[2], bfrag[4];
        #pragma unroll
        for (int i = 0; i < 2; ++i) {
            int m = waveM * 32 + i * 16 + ln;
            v8bf lo = *(const v8bf*)&Ash[m][lg * 8];
            v8bf hi = *(const v8bf*)&Ash[m][16 + lg * 8];
            afrag[i] = combine16(lo, hi);
        }
        #pragma unroll
        for (int j = 0; j < 4; ++j) {
            int n = waveN * 64 + j * 16 + ln;
            v8bf lo = *(const v8bf*)&Bsh[n][lg * 8];
            v8bf hi = *(const v8bf*)&Bsh[n][16 + lg * 8];
            bfrag[j] = combine16(lo, hi);
        }
        #pragma unroll
        for (int i = 0; i < 2; ++i)
            #pragma unroll
            for (int j = 0; j < 4; ++j)
                acc[i][j] = __builtin_amdgcn_wmma_f32_16x16x32_bf16(
                    false, afrag[i], false, bfrag[j], (short)0, acc[i][j],
                    false, false);
    }

    // ---- epilogue: C frag layout: VGPR r -> M = r + 8*lg, N = ln ----
    #pragma unroll
    for (int i = 0; i < 2; ++i) {
        #pragma unroll
        for (int j = 0; j < 4; ++j) {
            #pragma unroll
            for (int r = 0; r < 8; ++r) {
                int row = m0 + waveM * 32 + i * 16 + r + lg * 8;
                int col = n0 + waveN * 64 + j * 16 + ln;
                float v = acc[i][j][r];
                if (HAS_BIAS) v += bias[col];
                if (OUT_BF16)
                    ((bf16_t*)Cptr)[(size_t)row * N + col] = (bf16_t)v;
                else
                    ((float*)Cptr)[(size_t)row * N + col] = v;
            }
        }
    }
}

// ---------------------------------------------------------------------------
// Flash attention: block = (b, h, 64-query tile), 128 threads = 4 waves.
// Wave w owns query rows [qt*64 + w*16, +16): online softmax state (m,l) per
// row lives in registers (replicated across each 16-lane half).
// qkv layout [B*T][3C], split order k | q | v (per reference).
// K tile staged by the Tensor Data Mover; V staged transposed by VALU.
// ---------------------------------------------------------------------------
#define KOFF 0                 // Ksh: [key][80]      (64*80 bf16)
#define VOFF (64 * 80)         // Vsh: [d][80] keys   (64*80 bf16, transposed)
#define POFF (2 * 64 * 80)     // Psh: [wave][16][80]

__global__ __launch_bounds__(128) void attn_flash_k(
    const bf16_t* __restrict__ qkv, bf16_t* __restrict__ y)
{
    __shared__ __align__(16) bf16_t smem[2 * 64 * 80 + 4 * 16 * 80];

    const int tid  = threadIdx.x;
    const int wave = tid >> 5;
    const int lane = tid & 31;
    const int ln   = lane & 15;
    const int lg   = lane >> 4;

    const int qt = blockIdx.x;            // 0..31  (64-row query tile)
    const int bh = blockIdx.y;            // 0..31
    const int b  = bh / Hn;
    const int h  = bh % Hn;

    const size_t rstride = 3 * Cn;        // qkv row stride (elements)
    const bf16_t* kbase = qkv + (size_t)b * Tn * rstride + 0 * Cn + h * HDn;
    const bf16_t* qbase = qkv + (size_t)b * Tn * rstride + 1 * Cn + h * HDn;
    const bf16_t* vbase = qkv + (size_t)b * Tn * rstride + 2 * Cn + h * HDn;

    // ---- Q fragments (kept in registers): rows qt*64 + wave*16 + ln ----
    v16bf qfrag[2];
    {
        const size_t qrow = (size_t)qt * 64 + wave * 16 + ln;
        #pragma unroll
        for (int s = 0; s < 2; ++s) {
            v8bf lo = *(const v8bf*)(qbase + qrow * rstride + s * 32 + lg * 8);
            v8bf hi = *(const v8bf*)(qbase + qrow * rstride + s * 32 + 16 + lg * 8);
            qfrag[s] = combine16(lo, hi);
        }
    }

    v8f o[4] = {};
    float mrun[8], lrun[8];
    #pragma unroll
    for (int r = 0; r < 8; ++r) { mrun[r] = -1e30f; lrun[r] = 0.0f; }
    const float scale = 0.125f;           // 1/sqrt(64)

    for (int j = 0; j <= qt; ++j) {
        __syncthreads();                  // previous tiles fully consumed
        // ---- K tile via Tensor Data Mover (wave 0 issues + waits) ----
        if (wave == 0) {
            tdm_load_k_tile(KOFF * 2 /*bytes*/,
                            kbase + (size_t)(j * 64) * rstride,
                            (u32)(rstride / 2) /* DWORD stride */);
        }
        // ---- V tile staged transposed: Vsh[d][key] ----
        #pragma unroll
        for (int p = 0; p < 4; ++p) {
            int key = (tid >> 3) + p * 16;         // 0..63
            int dd  = (tid & 7) * 8;               // 0..56
            v8bf vv = *(const v8bf*)(vbase + (size_t)(j * 64 + key) * rstride + dd);
            #pragma unroll
            for (int i = 0; i < 8; ++i) smem[VOFF + (dd + i) * 80 + key] = vv[i];
        }
        if (wave == 0) __builtin_amdgcn_s_wait_tensorcnt(0);
        __syncthreads();                  // publish K (TDM) + V to all waves

        // ---- S = scale * Q K^T  (16 x 64 per wave) ----
        v8f s[4];
        #pragma unroll
        for (int t = 0; t < 4; ++t) {
            v8f a = {};
            #pragma unroll
            for (int ks = 0; ks < 2; ++ks) {
                int key = t * 16 + ln;
                v8bf lo = *(const v8bf*)&smem[KOFF + key * 80 + ks * 32 + lg * 8];
                v8bf hi = *(const v8bf*)&smem[KOFF + key * 80 + ks * 32 + 16 + lg * 8];
                a = __builtin_amdgcn_wmma_f32_16x16x32_bf16(
                        false, qfrag[ks], false, combine16(lo, hi),
                        (short)0, a, false, false);
            }
            s[t] = a;
        }
        #pragma unroll
        for (int t = 0; t < 4; ++t)
            #pragma unroll
            for (int r = 0; r < 8; ++r)
                s[t][r] *= scale;

        // ---- causal mask (only the diagonal block) ----
        if (j == qt) {
            #pragma unroll
            for (int t = 0; t < 4; ++t) {
                int key = t * 16 + ln;
                #pragma unroll
                for (int r = 0; r < 8; ++r) {
                    int qr = wave * 16 + r + lg * 8;
                    if (key > qr) s[t][r] = -1e30f;
                }
            }
        }

        // ---- online softmax per row (row = r + 8*lg within wave) ----
        float fscl[8];
        #pragma unroll
        for (int r = 0; r < 8; ++r) {
            float mt = fmaxf(fmaxf(s[0][r], s[1][r]), fmaxf(s[2][r], s[3][r]));
            #pragma unroll
            for (int off = 8; off > 0; off >>= 1)
                mt = fmaxf(mt, __shfl_xor(mt, off, 16));
            float mn = fmaxf(mrun[r], mt);
            float f  = __expf(mrun[r] - mn);
            float lt = 0.0f;
            #pragma unroll
            for (int t = 0; t < 4; ++t) {
                float p = __expf(s[t][r] - mn);
                s[t][r] = p;
                lt += p;
            }
            #pragma unroll
            for (int off = 8; off > 0; off >>= 1)
                lt += __shfl_xor(lt, off, 16);
            lrun[r] = lrun[r] * f + lt;
            mrun[r] = mn;
            fscl[r] = f;
        }
        #pragma unroll
        for (int t = 0; t < 4; ++t)
            #pragma unroll
            for (int r = 0; r < 8; ++r)
                o[t][r] *= fscl[r];

        // ---- P: C-frag layout -> LDS -> A-frag layout (per-wave region,
        //      same-wave LDS ops are in-order; no barrier needed) ----
        #pragma unroll
        for (int t = 0; t < 4; ++t)
            #pragma unroll
            for (int r = 0; r < 8; ++r)
                smem[POFF + wave * 1280 + (r + lg * 8) * 80 + t * 16 + ln] =
                    (bf16_t)s[t][r];

        // ---- O += P @ V ----
        #pragma unroll
        for (int ks = 0; ks < 2; ++ks) {
            const bf16_t* pw = &smem[POFF + wave * 1280 + ln * 80];
            v8bf plo = *(const v8bf*)(pw + ks * 32 + lg * 8);
            v8bf phi = *(const v8bf*)(pw + ks * 32 + 16 + lg * 8);
            v16bf pfrag = combine16(plo, phi);
            #pragma unroll
            for (int t = 0; t < 4; ++t) {
                int d = t * 16 + ln;
                v8bf lo = *(const v8bf*)&smem[VOFF + d * 80 + ks * 32 + lg * 8];
                v8bf hi = *(const v8bf*)&smem[VOFF + d * 80 + ks * 32 + 16 + lg * 8];
                o[t] = __builtin_amdgcn_wmma_f32_16x16x32_bf16(
                           false, pfrag, false, combine16(lo, hi),
                           (short)0, o[t], false, false);
            }
        }
    }

    // ---- normalize and write y[b, q, h*64 + d] (bf16) ----
    #pragma unroll
    for (int t = 0; t < 4; ++t) {
        #pragma unroll
        for (int r = 0; r < 8; ++r) {
            int q   = qt * 64 + wave * 16 + r + lg * 8;
            int col = h * HDn + t * 16 + ln;
            float v = o[t][r] / lrun[r];
            y[((size_t)b * Tn + q) * Cn + col] = (bf16_t)v;
        }
    }
}

// ---------------------------------------------------------------------------
extern "C" void kernel_launch(void* const* d_in, const int* in_sizes, int n_in,
                              void* d_out, int out_size, void* d_ws, size_t ws_size,
                              hipStream_t stream) {
    const float* x      = (const float*)d_in[0];   // [B,T,C]
    const float* w_attn = (const float*)d_in[1];   // [C,3C]
    const float* w_proj = (const float*)d_in[2];   // [C,C]
    const float* b_proj = (const float*)d_in[3];   // [C]
    float* out = (float*)d_out;                    // [B,T,C]

    const int M = Bn * Tn;                         // 4096
    bf16_t* qkv = (bf16_t*)d_ws;                   // [M][3C] bf16
    bf16_t* y   = qkv + (size_t)M * 3 * Cn;        // [M][C]  bf16

    // 1) qkv = x @ w_attn   (M=4096, N=3072, K=1024), bf16 out to ws
    gemm_wmma_k<0, 0, 1><<<dim3(3 * Cn / 128, M / 128), 256, 0, stream>>>(
        x, w_attn, nullptr, qkv, M, 3 * Cn, Cn);

    // 2) flash attention -> y (bf16)
    attn_flash_k<<<dim3(Tn / 64, Bn * Hn), 128, 0, stream>>>(qkv, y);

    // 3) out = y @ w_proj + b_proj  (M=4096, N=1024, K=1024), fp32 out
    gemm_wmma_k<1, 1, 0><<<dim3(Cn / 128, M / 128), 256, 0, stream>>>(
        y, w_proj, b_proj, out, M, Cn, Cn);
}
